// MultiHeadAttention_74698071212182
// MI455X (gfx1250) — matrix operand order repaired and checked by reading, not compiled, not run on hardware
//
#include <hip/hip_runtime.h>
#include <math.h>

typedef _Float16 h16;
typedef __attribute__((ext_vector_type(8)))  _Float16 v8h;
typedef __attribute__((ext_vector_type(16))) _Float16 v16h;
typedef __attribute__((ext_vector_type(4)))  float v4f;
typedef __attribute__((ext_vector_type(8)))  float v8f;

static constexpr int BSZ = 2, SEQ = 2048, DM = 1024, NH = 16, DH = 64;
static constexpr int MROWS = BSZ * SEQ;      // 4096
static constexpr float SM_SCALE = 0.125f;    // 1/sqrt(64)

__device__ __forceinline__ v8f wmma16(v16h a, v16h b, v8f c) {
  // D(16x16,f32) = A(16x32,f16) * B(32x16,f16) + C
  return __builtin_amdgcn_wmma_f32_16x16x32_f16(false, a, false, b, (short)0, c,
                                                false, false);
}
__device__ __forceinline__ v16h cat8(v8h lo, v8h hi) {
  return __builtin_shufflevector(lo, hi, 0, 1, 2, 3, 4, 5, 6, 7, 8, 9, 10, 11,
                                 12, 13, 14, 15);
}
__device__ __forceinline__ v8h pack8(v4f f0, v4f f1) {
  v8h h;
  h[0] = (h16)f0.x; h[1] = (h16)f0.y; h[2] = (h16)f0.z; h[3] = (h16)f0.w;
  h[4] = (h16)f1.x; h[5] = (h16)f1.y; h[6] = (h16)f1.z; h[7] = (h16)f1.w;
  return h;
}
__device__ __forceinline__ float redmax16(float v) {
  v = fmaxf(v, __shfl_xor(v, 1, 32));
  v = fmaxf(v, __shfl_xor(v, 2, 32));
  v = fmaxf(v, __shfl_xor(v, 4, 32));
  v = fmaxf(v, __shfl_xor(v, 8, 32));
  return v;
}
__device__ __forceinline__ float redsum16(float v) {
  v += __shfl_xor(v, 1, 32);
  v += __shfl_xor(v, 2, 32);
  v += __shfl_xor(v, 4, 32);
  v += __shfl_xor(v, 8, 32);
  return v;
}

// ---------------------------------------------------------------------------
// GEMM C = A(MxK,f32) * B(KxN,f32) via V_WMMA_F32_16X16X32_F16.
// Block tile 128(M) x 64(N), K-step 64; A/B converted to f16 in LDS.
// MODE 0: C f32 row-major [M][N]            (output projection -> d_out)
// MODE 1: RoPE epilogue, f16 [B][H][S][DH]  (Q/K projections)
// MODE 2: f16 [B][H][DH][S] (transposed V)  (V projection)
// ---------------------------------------------------------------------------
template <int MODE>
__global__ __launch_bounds__(256) void gemm_wmma_k(
    const float* __restrict__ A, const float* __restrict__ B,
    void* __restrict__ Cout, const float* __restrict__ cosT,
    const float* __restrict__ sinT, int M, int N, int K) {
  __shared__ h16 Ash[128][72];  // [m][k], row stride 144B (16B aligned)
  __shared__ h16 Bsh[64][72];   // [n][k] transposed, row stride 144B

  const int t = threadIdx.x;
  const int lane = t & 31, wv = t >> 5;
  const int row16 = lane & 15, hi = lane >> 4;
  const int m0 = blockIdx.x * 128, n0 = blockIdx.y * 64;

  v8f zero;
#pragma unroll
  for (int i = 0; i < 8; ++i) zero[i] = 0.0f;
  v8f acc[4];
#pragma unroll
  for (int nt = 0; nt < 4; ++nt) acc[nt] = zero;

  for (int k0 = 0; k0 < K; k0 += 64) {
    // Stage A 128x64: thread t -> row t>>1, 32-wide half-row (t&1)
    {
      const int m = t >> 1, ko = (t & 1) * 32;
      const float* ap = &A[(size_t)(m0 + m) * K + k0 + ko];
#pragma unroll
      for (int j = 0; j < 4; ++j) {
        v4f f0 = *(const v4f*)&ap[8 * j];
        v4f f1 = *(const v4f*)&ap[8 * j + 4];
        *(v8h*)&Ash[m][ko + 8 * j] = pack8(f0, f1);
      }
    }
    // Stage B 64x64 transposed: thread t -> k row t>>2, 16 columns (t&3)*16
    {
      const int k = t >> 2, nb = (t & 3) * 16;
      const float* bp = &B[(size_t)(k0 + k) * N + n0 + nb];
#pragma unroll
      for (int j = 0; j < 4; ++j) {
        v4f f = *(const v4f*)&bp[4 * j];
        Bsh[nb + 4 * j + 0][k] = (h16)f.x;
        Bsh[nb + 4 * j + 1][k] = (h16)f.y;
        Bsh[nb + 4 * j + 2][k] = (h16)f.z;
        Bsh[nb + 4 * j + 3][k] = (h16)f.w;
      }
    }
    __syncthreads();

#pragma unroll
    for (int ksub = 0; ksub < 2; ++ksub) {
      v8h alo = *(const v8h*)&Ash[wv * 16 + row16][ksub * 32 + hi * 8];
      v8h ahi = *(const v8h*)&Ash[wv * 16 + row16][ksub * 32 + 16 + hi * 8];
      v16h a = cat8(alo, ahi);
#pragma unroll
      for (int nt = 0; nt < 4; ++nt) {
        v8h blo = *(const v8h*)&Bsh[nt * 16 + row16][ksub * 32 + hi * 8];
        v8h bhi = *(const v8h*)&Bsh[nt * 16 + row16][ksub * 32 + 16 + hi * 8];
        acc[nt] = wmma16(a, cat8(blo, bhi), acc[nt]);
      }
    }
    __syncthreads();
  }

  // Epilogue. C layout: lane -> column nt*16+row16, VGPR i -> row i+8*hi.
  if (MODE == 0) {
    float* C = (float*)Cout;
#pragma unroll
    for (int nt = 0; nt < 4; ++nt) {
#pragma unroll
      for (int i = 0; i < 8; ++i) {
        int m = m0 + wv * 16 + i + 8 * hi;
        int n = n0 + nt * 16 + row16;
        C[(size_t)m * N + n] = acc[nt][i];
      }
    }
  } else {
    h16* C = (h16*)Cout;
#pragma unroll
    for (int nt = 0; nt < 4; ++nt) {
#pragma unroll
      for (int i = 0; i < 8; ++i) {
        int m = m0 + wv * 16 + i + 8 * hi;
        int n = n0 + nt * 16 + row16;
        int b = m >> 11, s = m & (SEQ - 1);
        int h = n >> 6, dh = n & (DH - 1);
        float val = acc[nt][i];
        if (MODE == 1) {
          // RoPE: pair (2j,2j+1) sits in adjacent lanes (col parity==lane parity)
          int j = dh >> 1;
          float cv = cosT[s * (DH / 2) + j];
          float sv = sinT[s * (DH / 2) + j];
          float other = __shfl_xor(val, 1, 32);
          val = ((dh & 1) == 0) ? (val * cv - other * sv)
                                : (other * sv + val * cv);
          C[(size_t)((b * NH + h) * SEQ + s) * DH + dh] = (h16)val;
        } else {  // MODE 2: V transposed [B][H][DH][S]
          C[(size_t)((b * NH + h) * DH + dh) * SEQ + s] = (h16)val;
        }
      }
    }
  }
}

// ---------------------------------------------------------------------------
// Flash attention: grid (S/128, B*H), 8 waves, wave owns 16 query rows.
// Q,K f16 [bh][s][dh]; V f16 [bh][dh][s] -> all WMMA fragments are contiguous
// 16B runs read straight from global (L2-resident). Only P is staged through
// wave-private LDS (C-layout -> A-layout + f32->f16). No block barriers.
// ---------------------------------------------------------------------------
__global__ __launch_bounds__(256) void flash_attn_k(
    const h16* __restrict__ Q, const h16* __restrict__ Km,
    const h16* __restrict__ Vm, float* __restrict__ O) {
  __shared__ h16 Psh[8][16][72];  // per-wave [row][key], stride 144B

  const int t = threadIdx.x;
  const int lane = t & 31, wv = t >> 5;
  const int row16 = lane & 15, hi = lane >> 4;
  const int bh = blockIdx.y;
  const size_t base = (size_t)bh * SEQ * DH;
  const int q0 = blockIdx.x * 128 + wv * 16;

  v8f zero;
#pragma unroll
  for (int i = 0; i < 8; ++i) zero[i] = 0.0f;

  // Persistent Q A-fragments (Dh=64 -> 2 k-sub-steps of 32)
  v16h aq[2];
#pragma unroll
  for (int ksub = 0; ksub < 2; ++ksub) {
    const h16* qp = &Q[base + (size_t)(q0 + row16) * DH + ksub * 32];
    aq[ksub] = cat8(*(const v8h*)&qp[hi * 8], *(const v8h*)&qp[16 + hi * 8]);
  }

  v8f acc[4];
#pragma unroll
  for (int nt = 0; nt < 4; ++nt) acc[nt] = zero;
  float mrow[8], lrow[8];
#pragma unroll
  for (int i = 0; i < 8; ++i) {
    mrow[i] = -INFINITY;
    lrow[i] = 0.0f;
  }

  for (int kt = 0; kt < SEQ / 64; ++kt) {
    const int keyrow = kt * 64;

    // Scores S = Q * K^T (16x64): B[kdim=dh][col=key] = K[key][dh]
    v8f sc[4];
#pragma unroll
    for (int nt = 0; nt < 4; ++nt) sc[nt] = zero;
#pragma unroll
    for (int nt = 0; nt < 4; ++nt) {
      const h16* kp = &Km[base + (size_t)(keyrow + nt * 16 + row16) * DH];
#pragma unroll
      for (int ksub = 0; ksub < 2; ++ksub) {
        v8h blo = *(const v8h*)&kp[ksub * 32 + hi * 8];
        v8h bhi = *(const v8h*)&kp[ksub * 32 + 16 + hi * 8];
        sc[nt] = wmma16(aq[ksub], cat8(blo, bhi), sc[nt]);
      }
    }

    // Online softmax per row (row i+8*hi lives in one 16-lane half)
#pragma unroll
    for (int i = 0; i < 8; ++i) {
      float s0 = sc[0][i] * SM_SCALE, s1 = sc[1][i] * SM_SCALE;
      float s2 = sc[2][i] * SM_SCALE, s3 = sc[3][i] * SM_SCALE;
      float mx = redmax16(fmaxf(fmaxf(s0, s1), fmaxf(s2, s3)));
      float mnew = fmaxf(mrow[i], mx);
      float alpha = __expf(mrow[i] - mnew);
      mrow[i] = mnew;
      float p0 = __expf(s0 - mnew), p1 = __expf(s1 - mnew);
      float p2 = __expf(s2 - mnew), p3 = __expf(s3 - mnew);
      lrow[i] = lrow[i] * alpha + redsum16(p0 + p1 + p2 + p3);
#pragma unroll
      for (int nt = 0; nt < 4; ++nt) acc[nt][i] *= alpha;
      Psh[wv][i + 8 * hi][row16] = (h16)p0;
      Psh[wv][i + 8 * hi][16 + row16] = (h16)p1;
      Psh[wv][i + 8 * hi][32 + row16] = (h16)p2;
      Psh[wv][i + 8 * hi][48 + row16] = (h16)p3;
    }

    // acc += P(16x64) * V(64x64): B[kdim=key][col=dh] = V^T[dh][key]
#pragma unroll
    for (int ksub = 0; ksub < 2; ++ksub) {
      v8h alo = *(const v8h*)&Psh[wv][row16][ksub * 32 + hi * 8];
      v8h ahi = *(const v8h*)&Psh[wv][row16][ksub * 32 + 16 + hi * 8];
      v16h a = cat8(alo, ahi);
#pragma unroll
      for (int nt = 0; nt < 4; ++nt) {
        const h16* vp = &Vm[base + (size_t)(nt * 16 + row16) * SEQ + keyrow +
                            ksub * 32 + hi * 8];
        acc[nt] = wmma16(a, cat8(*(const v8h*)vp, *(const v8h*)(vp + 16)),
                         acc[nt]);
      }
    }
  }

  // Normalize, write O f32 in [B][S][H*DH] for the output projection
  const int b = bh >> 4, h = bh & (NH - 1);
#pragma unroll
  for (int nt = 0; nt < 4; ++nt) {
#pragma unroll
    for (int i = 0; i < 8; ++i) {
      int s = q0 + i + 8 * hi;
      int dh = nt * 16 + row16;
      O[((size_t)(b * SEQ + s)) * (NH * DH) + h * DH + dh] = acc[nt][i] / lrow[i];
    }
  }
}

extern "C" void kernel_launch(void* const* d_in, const int* in_sizes, int n_in,
                              void* d_out, int out_size, void* d_ws,
                              size_t ws_size, hipStream_t stream) {
  (void)in_sizes; (void)n_in; (void)out_size; (void)ws_size;
  const float* x    = (const float*)d_in[0];
  const float* cosT = (const float*)d_in[1];
  const float* sinT = (const float*)d_in[2];
  const float* wq   = (const float*)d_in[3];
  const float* wk   = (const float*)d_in[4];
  const float* wvv  = (const float*)d_in[5];
  const float* wo   = (const float*)d_in[6];
  float* out = (float*)d_out;

  char* wsb = (char*)d_ws;
  const size_t HSZ = (size_t)MROWS * DM * sizeof(h16);  // 8 MiB per f16 buffer
  h16* Qt = (h16*)(wsb);
  h16* Kt = (h16*)(wsb + HSZ);
  h16* Vt = (h16*)(wsb + 2 * HSZ);
  float* Ot = (float*)(wsb + 3 * HSZ);  // 16 MiB f32

  dim3 blk(256);
  dim3 gproj(MROWS / 128, DM / 64);  // (32, 16)

  gemm_wmma_k<1><<<gproj, blk, 0, stream>>>(x, wq, Qt, cosT, sinT, MROWS, DM, DM);
  gemm_wmma_k<1><<<gproj, blk, 0, stream>>>(x, wk, Kt, cosT, sinT, MROWS, DM, DM);
  gemm_wmma_k<2><<<gproj, blk, 0, stream>>>(x, wvv, Vt, nullptr, nullptr, MROWS, DM, DM);

  flash_attn_k<<<dim3(SEQ / 128, BSZ * NH), blk, 0, stream>>>(Qt, Kt, Vt, Ot);

  gemm_wmma_k<0><<<gproj, blk, 0, stream>>>(Ot, wo, out, nullptr, nullptr, MROWS, DM, DM);
}
